// GATLayer_10393820856962
// MI455X (gfx1250) — matrix operand order repaired and checked by reading, not compiled, not run on hardware
//
#include <hip/hip_runtime.h>

typedef __attribute__((ext_vector_type(16))) _Float16 v16h;
typedef __attribute__((ext_vector_type(8)))  _Float16 v8h;
typedef __attribute__((ext_vector_type(8)))  float    v8f;
typedef __attribute__((ext_vector_type(4)))  unsigned int u32x4;
typedef __attribute__((ext_vector_type(8)))  int i32x8;
typedef __attribute__((ext_vector_type(4)))  int i32x4;

#define B_    4
#define N_    2048
#define FI_   512
#define FO_   64
#define H_    8
#define HF_   (H_ * FO_)        // 512
#define NEG_INF_ (-9.0e15f)
#define ALPHA_   0.2f
#define KSTAGE   256
#define NSTAGE   (N_ / KSTAGE)  // 8

#if defined(__AMDGCN__) && __has_builtin(__builtin_amdgcn_tensor_load_to_lds)
#define USE_TDM 1
#else
#define USE_TDM 0
#endif

// A-operand K map (16x32 f16 A): lanes 0-15 hold K=0..15, lanes 16-31 K=16..31
// (same contiguous order as bkmap); halves are consecutive K pairs.
__device__ __forceinline__ int akmap(int t, int hi) {
    return t + 16 * hi;
}
// B-operand (32x16 f16): lanes 0-15 hold K=0..15 across halves, lanes 16-31 K=16..31
__device__ __forceinline__ int bkmap(int t, int hi) {
    return t + 16 * hi;
}

// ---------------------------------------------------------------------------
// Pack x (f32) into f16 A-fragment order:
//   xp[((b*128+rt)*16 + kc)*512 + lane*16 + t] = x[b][rt*16+(lane&15)][kc*32+akmap(t,lane>>4)]
// ---------------------------------------------------------------------------
__global__ __launch_bounds__(256)
void gat_pack_x(const float* __restrict__ x, _Float16* __restrict__ xp) {
    int idx  = blockIdx.x * 256 + threadIdx.x;   // over B*N*FI = 4M
    int t    = idx & 15;
    int lane = (idx >> 4) & 31;
    int kc   = (idx >> 9) & 15;                  // FI/32
    int rt   = (idx >> 13) & 127;                // N/16
    int b    = idx >> 20;
    int hi = lane >> 4, ln = lane & 15;
    int row = rt * 16 + ln;
    int k   = kc * 32 + akmap(t, hi);
    xp[idx] = (_Float16)x[((size_t)b * N_ + row) * FI_ + k];
}

// ---------------------------------------------------------------------------
// Pack W (f32) into f16 B-fragment order:
//   wp[(((head*16+kc)*4+nt)*32 + lane)*16 + t] = W[head][kc*32+bkmap(t,hi)][nt*16+ln]
// ---------------------------------------------------------------------------
__global__ __launch_bounds__(256)
void gat_pack_w(const float* __restrict__ W, _Float16* __restrict__ wp) {
    int idx  = blockIdx.x * 256 + threadIdx.x;   // over H*FI*FO = 256K
    int t    = idx & 15;
    int lane = (idx >> 4) & 31;
    int nt   = (idx >> 9) & 3;
    int kc   = (idx >> 11) & 15;
    int head = idx >> 15;
    int hi = lane >> 4, ln = lane & 15;
    int k = kc * 32 + bkmap(t, hi);
    int n = nt * 16 + ln;
    wp[idx] = (_Float16)W[((size_t)head * FI_ + k) * FO_ + n];
}

// ---------------------------------------------------------------------------
// Pack adjacency int32 -> bitmask (wave32 ballot), 67MB -> 2MB
// ---------------------------------------------------------------------------
__global__ __launch_bounds__(256)
void gat_maskpack(const int* __restrict__ adj, unsigned* __restrict__ mbits) {
    int gid = blockIdx.x * 256 + threadIdx.x;    // over B*N*N
    int pred = adj[gid] > 0;
    unsigned long long bal = __ballot(pred);     // wave32: low 32 bits valid
    if ((threadIdx.x & 31) == 0) mbits[gid >> 5] = (unsigned)bal;
}

// ---------------------------------------------------------------------------
// Projection: h[b,h] = x[b] @ W[h] from packed fragments; writes h in packed
// B-fragment order for the attention matmul, plus s1/s2 row scores via
// half-wave shfl reductions.
// One wave = 16 rows x 64 cols; workgroup = 8 waves = 128 rows of one (b,head)
// ---------------------------------------------------------------------------
__global__ __launch_bounds__(256)
void gat_proj(const _Float16* __restrict__ xp, const _Float16* __restrict__ wp,
              const float* __restrict__ a1, const float* __restrict__ a2,
              _Float16* __restrict__ hp, float* __restrict__ s1,
              float* __restrict__ s2) {
    const int wave = threadIdx.x >> 5;
    const int lane = threadIdx.x & 31;
    const int hi   = lane >> 4;
    const int ln   = lane & 15;

    int blk    = blockIdx.x;
    int rowblk = blk % (N_ / 128);
    int head   = (blk / (N_ / 128)) % H_;
    int b      = blk / ((N_ / 128) * H_);
    int i0     = rowblk * 128 + wave * 16;
    int rt     = i0 >> 4;

    const _Float16* xpb = xp + ((size_t)b * 128 + rt) * 16 * 512;
    const _Float16* wph = wp + (size_t)head * 16 * 4 * 32 * 16;

    v8f acc[4] = {};
    for (int kc = 0; kc < 16; ++kc) {
        v16h a = *(const v16h*)(xpb + (size_t)kc * 512 + lane * 16);
        #pragma unroll
        for (int nt = 0; nt < 4; ++nt) {
            v16h bf = *(const v16h*)(wph + (((size_t)kc * 4 + nt) * 32 + lane) * 16);
            acc[nt] = __builtin_amdgcn_wmma_f32_16x16x32_f16(
                false, a, false, bf, (short)0, acc[nt], false, false);
        }
    }

    // ---- write h in packed B-fragment order (each lane: 8 contiguous halves)
    int kc0   = i0 >> 5;
    int krel0 = (i0 & 31) + 8 * hi;        // i0&31 in {0,16}
    int hi2   = krel0 >> 4;
    int t0    = krel0 & 15;                // 0 or 8 -> 16B aligned
    _Float16* hpb = hp + (size_t)(b * H_ + head) * N_ * FO_;
    #pragma unroll
    for (int nt = 0; nt < 4; ++nt) {
        v8h h8;
        #pragma unroll
        for (int v = 0; v < 8; ++v) h8[v] = (_Float16)acc[nt][v];
        size_t idx = (((size_t)kc0 * 4 + nt) * 32 + hi2 * 16 + ln) * 16 + t0;
        *(v8h*)(hpb + idx) = h8;
    }

    // ---- s1/s2: per-row dot with a1/a2, reduced across the 16-lane halves
    float r1[8] = {}, r2[8] = {};
    #pragma unroll
    for (int nt = 0; nt < 4; ++nt) {
        float a1v = a1[head * FO_ + nt * 16 + ln];
        float a2v = a2[head * FO_ + nt * 16 + ln];
        #pragma unroll
        for (int v = 0; v < 8; ++v) {
            r1[v] += acc[nt][v] * a1v;
            r2[v] += acc[nt][v] * a2v;
        }
    }
    size_t sbase = (size_t)(b * H_ + head) * N_;
    #pragma unroll
    for (int v = 0; v < 8; ++v) {
        float t1 = r1[v], t2 = r2[v];
        #pragma unroll
        for (int d = 1; d < 16; d <<= 1) {
            t1 += __shfl_xor(t1, d, 32);
            t2 += __shfl_xor(t2, d, 32);
        }
        if (ln == 0) {
            s1[sbase + i0 + v + 8 * hi] = t1;
            s2[sbase + i0 + v + 8 * hi] = t2;
        }
    }
}

// ---------------------------------------------------------------------------
// TDM: DMA one 32KB h stage (contiguous) into LDS. D# per ISA ch8:
// group0 = {count=1, lds_addr, global_addr lo, global_addr hi | type=2}
// group1 = {data_size=2B, tensor_dim0=1M, tensor_dim1=1, tile_dim0=16384}
// 6-arg toolchain variant: (g0, g1, g2, g3, g4, cpol)
// ---------------------------------------------------------------------------
#if USE_TDM
__device__ __forceinline__ void tdm_issue(const _Float16* gsrc, void* ldst) {
    unsigned long long ga = (unsigned long long)(const void*)gsrc;
    unsigned ldso = (unsigned)(unsigned long long)(size_t)ldst;  // LDS byte offset
    u32x4 g0;
    g0[0] = 1u;                                            // count=1
    g0[1] = ldso;                                          // lds_addr
    g0[2] = (unsigned)ga;                                  // global_addr[31:0]
    g0[3] = ((unsigned)(ga >> 32) & 0x01FFFFFFu) | 0x80000000u; // [56:32] + type=2
    i32x8 g1;
    g1[0] = 0x00010000;          // workgroup_mask=0, data_size=1 (2 bytes)
    g1[1] = 0;                   // tensor_dim0 low16 = 0 (td0 = 1<<20)
    g1[2] = 0x00010010;          // td0 hi16 = 0x10, tensor_dim1 = 1
    g1[3] = (int)0x40000000;     // tile_dim0 = 16384 elements (32KB)
    g1[4] = 0;                   // tile_dim1/2 = 0 (unused)
    g1[5] = 0x00100000;          // tensor_dim0_stride = 1<<20
    g1[6] = 0;
    g1[7] = 0;
    i32x4 z4 = {0, 0, 0, 0};
    i32x8 z8 = {0, 0, 0, 0, 0, 0, 0, 0};
    __builtin_amdgcn_tensor_load_to_lds(g0, g1, z4, z4, z8, 0);
}
#endif

// ---------------------------------------------------------------------------
// Fused masked-softmax attention + p@h (flash-style, online softmax).
// Workgroup = (b, head, 128-row block); wave = one 16-row M tile, 64 out cols.
// h (packed fragments) staged via TDM double-buffer; scores from LDS s2.
// ---------------------------------------------------------------------------
__global__ __launch_bounds__(256)
void gat_attn(const _Float16* __restrict__ hp, const float* __restrict__ s1g,
              const float* __restrict__ s2g, const unsigned* __restrict__ mbits,
              float* __restrict__ out) {
    __shared__ __align__(16) _Float16 ldsH[2][KSTAGE * FO_];   // 2 x 32KB
    __shared__ __align__(16) float    ldsS2[N_];               // 8KB

    const int lane = threadIdx.x & 31;
    const int hi   = lane >> 4;
    const int ln   = lane & 15;
    const int wave = threadIdx.x >> 5;

    int blk    = blockIdx.x;
    int rowblk = blk % (N_ / 128);
    int head   = (blk / (N_ / 128)) % H_;
    int b      = blk / ((N_ / 128) * H_);
    int i0     = rowblk * 128 + wave * 16;
    int row    = i0 + ln;

    const _Float16* hbh  = hp + (size_t)(b * H_ + head) * N_ * FO_;  // packed
    const float*    s2h  = s2g + (size_t)(b * H_ + head) * N_;
    const unsigned* mrow = mbits + ((size_t)b * N_ + row) * (N_ / 32);
    const float     s1v  = s1g[(size_t)(b * H_ + head) * N_ + row];

    // stage all of s2 for this (b,head) once
    for (int j = threadIdx.x; j < N_; j += 256) ldsS2[j] = s2h[j];
#if USE_TDM
    if (threadIdx.x < 32) tdm_issue(hbh, &ldsH[0][0]);
#endif

    v8f   acc[4] = {};
    float m_run  = -3.0e38f;
    float l_run  = 0.0f;

    for (int s = 0; s < NSTAGE; ++s) {
        const _Float16* buf;
#if USE_TDM
        if (threadIdx.x < 32) {
            if (s + 1 < NSTAGE) {           // overlap next DMA with compute
                tdm_issue(hbh + (size_t)(s + 1) * KSTAGE * FO_,
                          &ldsH[(s + 1) & 1][0]);
                __builtin_amdgcn_s_wait_tensorcnt(1);
            } else {
                __builtin_amdgcn_s_wait_tensorcnt(0);
            }
        }
        __syncthreads();
        buf = &ldsH[s & 1][0];
#else
        {
            const uint4* src = (const uint4*)(hbh + (size_t)s * KSTAGE * FO_);
            uint4*       dst = (uint4*)&ldsH[0][0];
            for (int idx = threadIdx.x; idx < KSTAGE * FO_ / 8; idx += 256)
                dst[idx] = src[idx];
        }
        __syncthreads();
        buf = &ldsH[0][0];
#endif

        for (int sub = 0; sub < KSTAGE / 32; ++sub) {
            int j0 = s * KSTAGE + sub * 32;
            unsigned mword = mrow[j0 >> 5];

            float ev[16];
            float cmax = -3.0e38f;
            #pragma unroll
            for (int v = 0; v < 8; ++v) {
                int k = akmap(2 * v, hi);            // even; pair (k, k+1)
                float2 s2v = *(const float2*)&ldsS2[j0 + k];
                float e0 = s1v + s2v.x;
                float e1 = s1v + s2v.y;
                e0 = fmaxf(e0, ALPHA_ * e0);         // leakyrelu (slope<1)
                e1 = fmaxf(e1, ALPHA_ * e1);
                e0 = ((mword >> k) & 1u) ? e0 : NEG_INF_;
                e1 = ((mword >> (k + 1)) & 1u) ? e1 : NEG_INF_;
                ev[2 * v]     = e0;
                ev[2 * v + 1] = e1;
                cmax = fmaxf(cmax, fmaxf(e0, e1));
            }
            cmax = fmaxf(cmax, __shfl_xor(cmax, 16, 32));   // combine lane pair
            float m_new = fmaxf(m_run, cmax);

            float csum = 0.0f;
            v16h  a;
            #pragma unroll
            for (int t = 0; t < 16; ++t) {
                float p = __expf(ev[t] - m_new);
                csum += p;
                a[t] = (_Float16)p;
            }
            csum += __shfl_xor(csum, 16, 32);
            float corr = __expf(m_run - m_new);
            l_run = l_run * corr + csum;
            m_run = m_new;

            float cv[8];
            #pragma unroll
            for (int v = 0; v < 8; ++v)
                cv[v] = __shfl(corr, v + 8 * hi, 32);

            #pragma unroll
            for (int nt = 0; nt < 4; ++nt) {
                v16h bf = *(const v16h*)(buf + ((sub * 4 + nt) * 32 + lane) * 16);
                v8f c = acc[nt];
                #pragma unroll
                for (int v = 0; v < 8; ++v) c[v] *= cv[v];
                acc[nt] = __builtin_amdgcn_wmma_f32_16x16x32_f16(
                    false, a, false, bf, (short)0, c, false, false);
            }
        }
        __syncthreads();
    }

    // finalize: divide by softmax denominator (broadcast per C-row)
    float linv = 1.0f / l_run;
    float lv[8];
    #pragma unroll
    for (int v = 0; v < 8; ++v)
        lv[v] = __shfl(linv, v + 8 * hi, 32);

    float* outb = out + (size_t)b * N_ * HF_;
    #pragma unroll
    for (int nt = 0; nt < 4; ++nt)
        #pragma unroll
        for (int v = 0; v < 8; ++v) {
            int r = i0 + v + 8 * hi;
            outb[(size_t)r * HF_ + head * FO_ + nt * 16 + ln] =
                acc[nt][v] * lv[v];
        }
}

// ---------------------------------------------------------------------------
extern "C" void kernel_launch(void* const* d_in, const int* in_sizes, int n_in,
                              void* d_out, int out_size, void* d_ws,
                              size_t ws_size, hipStream_t stream) {
    const float* x   = (const float*)d_in[0];
    const int*   adj = (const int*)d_in[1];
    const float* W   = (const float*)d_in[2];
    const float* a1  = (const float*)d_in[3];
    const float* a2  = (const float*)d_in[4];
    float*       out = (float*)d_out;

    char* ws = (char*)d_ws;
    _Float16* xp = (_Float16*)ws;                       // 8.39 MB
    size_t off = (size_t)B_ * N_ * FI_ * sizeof(_Float16);
    _Float16* wpk = (_Float16*)(ws + off);              // 0.5 MB
    off += (size_t)H_ * FI_ * FO_ * sizeof(_Float16);
    _Float16* hp = (_Float16*)(ws + off);               // 8.39 MB
    off += (size_t)B_ * H_ * N_ * FO_ * sizeof(_Float16);
    float* s1 = (float*)(ws + off);
    off += (size_t)B_ * H_ * N_ * sizeof(float);
    float* s2 = (float*)(ws + off);
    off += (size_t)B_ * H_ * N_ * sizeof(float);
    unsigned* mbits = (unsigned*)(ws + off);            // 2.10 MB

    gat_pack_x<<<(B_ * N_ * FI_) / 256, 256, 0, stream>>>(x, xp);
    gat_pack_w<<<(H_ * FI_ * FO_) / 256, 256, 0, stream>>>(W, wpk);
    gat_maskpack<<<(B_ * N_ * N_) / 256, 256, 0, stream>>>(adj, mbits);
    gat_proj<<<B_ * H_ * (N_ / 128), 256, 0, stream>>>(xp, wpk, a1, a2, hp, s1, s2);
    gat_attn<<<B_ * H_ * (N_ / 128), 256, 0, stream>>>(hp, s1, s2, mbits, out);
}
